// BasicBlock_5179730559086
// MI455X (gfx1250) — 8-bit matrix operand order repaired and checked by reading, compiled, not run on hardware
//
#include <hip/hip_runtime.h>
#include <hip/hip_bf16.h>

typedef __attribute__((ext_vector_type(8))) int v8i;

#define EPSV 1e-5f
#define NPIX 25088

#define WS_ACT1   ((size_t)0)
#define WS_ACT2   ((size_t)6422528)
#define WS_OUT1   ((size_t)12845056)
#define WS_PB1    ((size_t)38535168)
#define WS_PB2    ((size_t)39124992)
#define WS_S1     ((size_t)39190528)
#define WS_C1     (WS_S1 + 1024)
#define WS_S2     (WS_S1 + 2048)
#define WS_C2     (WS_S1 + 3072)
#define WS_ZBUF   (WS_S1 + 4096)

static __device__ __forceinline__ int sgn8(float v) {
    return (v > 0.f) ? 1 : ((v < 0.f) ? -1 : 0);
}

__global__ __launch_bounds__(256)
void prep_channels(const float* __restrict__ w1, const float* __restrict__ w2,
                   const float* __restrict__ g1, const float* __restrict__ be1,
                   const float* __restrict__ m1, const float* __restrict__ v1,
                   const float* __restrict__ g2, const float* __restrict__ be2,
                   const float* __restrict__ m2, const float* __restrict__ v2,
                   const float* __restrict__ loss,
                   float* __restrict__ s1, float* __restrict__ c1,
                   float* __restrict__ s2, float* __restrict__ c2,
                   float* __restrict__ zbuf, float* __restrict__ outLoss)
{
    const int o = threadIdx.x;
    const float* p = w1 + (size_t)o * 2304;
    float sum1 = 0.f;
    for (int i = 0; i < 2304; ++i) sum1 += fabsf(p[i]);
    const float* q = w2 + (size_t)o * 256;
    float sum2 = 0.f;
    for (int i = 0; i < 256; ++i) sum2 += fabsf(q[i]);
    const float inv1 = g1[o] / sqrtf(v1[o] + EPSV);
    const float inv2 = g2[o] / sqrtf(v2[o] + EPSV);
    s1[o] = (sum1 / 2304.f) * inv1;
    c1[o] = be1[o] - m1[o] * inv1;
    s2[o] = (sum2 / 256.f) * inv2;
    c2[o] = be2[o] - m2[o] * inv2;
    if (o < 64) zbuf[o] = 0.f;
    if (o == 0) *outLoss = *loss;
}

__global__ __launch_bounds__(256)
void pack_weights(const float* __restrict__ w1, const float* __restrict__ w2,
                  signed char* __restrict__ pB1, signed char* __restrict__ pB2)
{
    int id = blockIdx.x * 256 + threadIdx.x;
    bool conv1;
    unsigned* dst;
    int t;
    if (id < 147456) { conv1 = true;  t = id;          dst = (unsigned*)pB1 + t; }
    else             { conv1 = false; t = id - 147456; dst = (unsigned*)pB2 + t; }
    const int kblk = t / 4096;  t %= 4096;
    const int nb   = t / 256;   t %= 256;
    const int lane = t / 8;
    const int jj   = t % 8;
    const int n  = nb * 16 + (lane & 15);
    const int lh = lane >> 4;
    unsigned dw = 0;
#pragma unroll
    for (int bb = 0; bb < 4; ++bb) {
        const int k_hw = ((jj >> 2) * 32) + lh * 16 + ((jj & 3) * 4) + bb;
        const int off  = k_hw & 3;
        const int kb   = k_hw & ~3;
        const int half = (kb >> 3) & 1;
        const int k2   = kb - half * 8;
        const int jA   = ((k2 >> 4) << 1) | ((k2 >> 2) & 1);
        const int c64  = lh * 32 + jj * 4 + bb;
        float wvf;
        if (conv1) {
            const int tapi = kblk >> 2;
            const int cin  = (kblk & 3) * 64 + c64;
            wvf = w1[((size_t)n * 256 + cin) * 9 + tapi];
        } else {
            const int cin = kblk * 64 + c64;
            wvf = w2[(size_t)n * 256 + cin];
        }
        dw |= ((unsigned)(sgn8(wvf) & 0xff)) << (8 * bb);
    }
    *dst = dw;
}

__global__ __launch_bounds__(256)
void act1_k(const float* __restrict__ x, const float* __restrict__ b11,
            signed char* __restrict__ act1)
{
    const int id  = blockIdx.x * 256 + threadIdx.x;
    const int pix = id >> 6;
    const int cq  = id & 63;
    const unsigned pp = (unsigned)pix;
    const int b  = pp / 784u;
    const int sp = pp % 784u;
    const float* xb = x + ((size_t)b * 256 + cq * 4) * 784 + sp;
    unsigned dw = 0;
#pragma unroll
    for (int i = 0; i < 4; ++i) {
        const float v = xb[(size_t)i * 784] + b11[cq * 4 + i];
        dw |= ((unsigned)(sgn8(v) & 0xff)) << (8 * i);
    }
    *(unsigned*)(act1 + (size_t)pix * 256 + cq * 4) = dw;
}

__global__ __launch_bounds__(256)
void conv3x3_wmma(const signed char* __restrict__ act,
                  const signed char* __restrict__ pB,
                  const float* __restrict__ x,
                  const float* __restrict__ s1, const float* __restrict__ c1,
                  const float* __restrict__ b12, const float* __restrict__ a1,
                  const float* __restrict__ b13, const float* __restrict__ b21,
                  float* __restrict__ out1,
                  signed char* __restrict__ act2,
                  const signed char* __restrict__ zbuf)
{
    const int lane  = threadIdx.x & 31;
    const int wv    = threadIdx.x >> 5;
    const int waveM = wv & 3;
    const int waveN = wv >> 2;
    const int mBase = blockIdx.x * 128 + waveM * 32;
    const int nBase = blockIdx.y * 128 + waveN * 64;
    const int lm = lane & 15;
    const int lh = lane >> 4;
    const int halfOff = lh * 32;

    const int pix0 = mBase + lm;
    const int pix1 = pix0 + 16;
    const unsigned up0 = (unsigned)pix0, up1 = (unsigned)pix1;
    const int b0 = up0 / 784u, sp0 = up0 % 784u;
    const int y0 = sp0 / 28,   x0c = sp0 % 28;
    const int b1 = up1 / 784u, sp1 = up1 % 784u;
    const int y1 = sp1 / 28,   x1c = sp1 % 28;

    v8i acc[2][4];
    const v8i vz = {0, 0, 0, 0, 0, 0, 0, 0};
#pragma unroll
    for (int i = 0; i < 2; ++i)
#pragma unroll
        for (int j = 0; j < 4; ++j) acc[i][j] = vz;

    const int nbG = nBase >> 4;

    auto tapBases = [&](int tap, const signed char*& b0o, const signed char*& b1o,
                        int& s0o, int& s1o) {
        const int dy = tap / 3 - 1, dx = tap % 3 - 1;
        const int ys0 = y0 + dy, xs0 = x0c + dx;
        const int ys1 = y1 + dy, xs1 = x1c + dx;
        const bool ok0 = ((unsigned)ys0 < 28u) && ((unsigned)xs0 < 28u);
        const bool ok1 = ((unsigned)ys1 < 28u) && ((unsigned)xs1 < 28u);
        b0o = ok0 ? act + ((size_t)(b0 * 784 + ys0 * 28 + xs0) * 256 + halfOff) : zbuf;
        b1o = ok1 ? act + ((size_t)(b1 * 784 + ys1 * 28 + xs1) * 256 + halfOff) : zbuf;
        s0o = ok0 ? 64 : 0;
        s1o = ok1 ? 64 : 0;
    };

    const signed char* base0; const signed char* base1; int st0, st1;
    tapBases(0, base0, base1, st0, st1);

    v8i aA[2][2], bB[2][4];
    aA[0][0] = *(const v8i*)(base0);
    aA[0][1] = *(const v8i*)(base1);
    {
        const signed char* pb0 = pB + ((size_t)nbG * 32 + lane) * 32;
#pragma unroll
        for (int nt = 0; nt < 4; ++nt)
            bB[0][nt] = *(const v8i*)(pb0 + (size_t)nt * 1024);
    }

    for (int tap = 0; tap < 9; ++tap) {
        const signed char* nbase0 = zbuf; const signed char* nbase1 = zbuf;
        int nst0 = 0, nst1 = 0;
        if (tap < 8) {
            tapBases(tap + 1, nbase0, nbase1, nst0, nst1);
            __builtin_prefetch(nbase0, 0, 1);
            __builtin_prefetch(nbase0 + 128, 0, 1);
            __builtin_prefetch(nbase1, 0, 1);
            __builtin_prefetch(nbase1 + 128, 0, 1);
        }
        const signed char* pb_tap =
            pB + (((size_t)tap * 64 + nbG) * 32 + lane) * 32;
#pragma unroll
        for (int cblk = 0; cblk < 4; ++cblk) {
            const int cur = cblk & 1;
            const int nxt = cur ^ 1;
            if (cblk < 3) {
                aA[nxt][0] = *(const v8i*)(base0 + (cblk + 1) * st0);
                aA[nxt][1] = *(const v8i*)(base1 + (cblk + 1) * st1);
                const signed char* pnb = pb_tap + (size_t)(cblk + 1) * 16384;
#pragma unroll
                for (int nt = 0; nt < 4; ++nt)
                    bB[nxt][nt] = *(const v8i*)(pnb + (size_t)nt * 1024);
            } else if (tap < 8) {
                aA[nxt][0] = *(const v8i*)(nbase0);
                aA[nxt][1] = *(const v8i*)(nbase1);
                const signed char* pnb =
                    pB + (((size_t)(tap + 1) * 64 + nbG) * 32 + lane) * 32;
#pragma unroll
                for (int nt = 0; nt < 4; ++nt)
                    bB[nxt][nt] = *(const v8i*)(pnb + (size_t)nt * 1024);
            }
#pragma unroll
            for (int nt = 0; nt < 4; ++nt) {
                acc[0][nt] = __builtin_amdgcn_wmma_i32_16x16x64_iu8(
                    true, aA[cur][0], true, bB[cur][nt], acc[0][nt], false, false);
                acc[1][nt] = __builtin_amdgcn_wmma_i32_16x16x64_iu8(
                    true, aA[cur][1], true, bB[cur][nt], acc[1][nt], false, false);
            }
        }
        base0 = nbase0; base1 = nbase1; st0 = nst0; st1 = nst1;
    }

#pragma unroll
    for (int nt = 0; nt < 4; ++nt) {
        const int nn = nBase + nt * 16 + lm;
        const float s   = s1[nn],  cc  = c1[nn];
        const float bb2 = b12[nn], aa  = a1[nn];
        const float bb3 = b13[nn], bb21 = b21[nn];
#pragma unroll
        for (int mt = 0; mt < 2; ++mt) {
#pragma unroll
            for (int r = 0; r < 8; ++r) {
                const int pix = mBase + mt * 16 + r + lh * 8;
                const unsigned pp = (unsigned)pix;
                const int bI = pp / 784u, rr = pp % 784u;
                const int yy = rr / 28,   xx = rr % 28;
                const float dot  = (float)acc[mt][nt][r];
                const float conv = dot * s + cc;
                const float xres = x[((size_t)(bI * 256 + nn)) * 784 + yy * 28 + xx];
                float t = conv + xres + bb2;
                t = (t > 0.f) ? t : aa * t;
                const float o1 = t + bb3;
                out1[(size_t)pix * 256 + nn] = o1;
                const float av = o1 + bb21;
                act2[(size_t)pix * 256 + nn] = (signed char)sgn8(av);
            }
        }
    }
}

__global__ __launch_bounds__(256)
void conv1x1_wmma(const signed char* __restrict__ act2,
                  const signed char* __restrict__ pB,
                  const float* __restrict__ out1,
                  const float* __restrict__ s2, const float* __restrict__ c2,
                  const float* __restrict__ b22, const float* __restrict__ a2,
                  const float* __restrict__ b23,
                  float* __restrict__ out)
{
    const int lane  = threadIdx.x & 31;
    const int wv    = threadIdx.x >> 5;
    const int waveM = wv & 3;
    const int waveN = wv >> 2;
    const int mBase = blockIdx.x * 128 + waveM * 32;
    const int nBase = blockIdx.y * 128 + waveN * 64;
    const int lm = lane & 15;
    const int lh = lane >> 4;
    const int halfOff = lh * 32;

    const int pix0 = mBase + lm;
    const int pix1 = pix0 + 16;

    v8i acc[2][4];
    const v8i vz = {0, 0, 0, 0, 0, 0, 0, 0};
#pragma unroll
    for (int i = 0; i < 2; ++i)
#pragma unroll
        for (int j = 0; j < 4; ++j) acc[i][j] = vz;

    const int nbG = nBase >> 4;
    const signed char* baseA0 = act2 + (size_t)pix0 * 256 + halfOff;
    const signed char* baseA1 = act2 + (size_t)pix1 * 256 + halfOff;
    const signed char* pb0 = pB + ((size_t)nbG * 32 + lane) * 32;

    v8i aA[2][2], bB[2][4];
    aA[0][0] = *(const v8i*)(baseA0);
    aA[0][1] = *(const v8i*)(baseA1);
#pragma unroll
    for (int nt = 0; nt < 4; ++nt)
        bB[0][nt] = *(const v8i*)(pb0 + (size_t)nt * 1024);

#pragma unroll
    for (int kblk = 0; kblk < 4; ++kblk) {
        const int cur = kblk & 1;
        const int nxt = cur ^ 1;
        if (kblk < 3) {
            aA[nxt][0] = *(const v8i*)(baseA0 + (kblk + 1) * 64);
            aA[nxt][1] = *(const v8i*)(baseA1 + (kblk + 1) * 64);
            const signed char* pnb = pb0 + (size_t)(kblk + 1) * 16384;
#pragma unroll
            for (int nt = 0; nt < 4; ++nt)
                bB[nxt][nt] = *(const v8i*)(pnb + (size_t)nt * 1024);
        }
#pragma unroll
        for (int nt = 0; nt < 4; ++nt) {
            acc[0][nt] = __builtin_amdgcn_wmma_i32_16x16x64_iu8(
                true, aA[cur][0], true, bB[cur][nt], acc[0][nt], false, false);
            acc[1][nt] = __builtin_amdgcn_wmma_i32_16x16x64_iu8(
                true, aA[cur][1], true, bB[cur][nt], acc[1][nt], false, false);
        }
    }

#pragma unroll
    for (int nt = 0; nt < 4; ++nt) {
        const int nn = nBase + nt * 16 + lm;
        const float s   = s2[nn],  cc = c2[nn];
        const float bb2 = b22[nn], aa = a2[nn];
        const float bb3 = b23[nn];
#pragma unroll
        for (int mt = 0; mt < 2; ++mt) {
#pragma unroll
            for (int r = 0; r < 8; ++r) {
                const int pix = mBase + mt * 16 + r + lh * 8;
                const unsigned pp = (unsigned)pix;
                const int bI = pp / 784u, rr = pp % 784u;
                const int yy = rr / 28,   xx = rr % 28;
                const float dot  = (float)acc[mt][nt][r];
                const float res  = out1[(size_t)pix * 256 + nn];
                float t = dot * s + cc + res + bb2;
                t = (t > 0.f) ? t : aa * t;
                out[((size_t)(bI * 256 + nn)) * 784 + yy * 28 + xx] = t + bb3;
            }
        }
    }
}

extern "C" void kernel_launch(void* const* d_in, const int* in_sizes, int n_in,
                              void* d_out, int out_size, void* d_ws, size_t ws_size,
                              hipStream_t stream)
{
    (void)in_sizes; (void)n_in; (void)out_size; (void)ws_size;
    const float* x    = (const float*)d_in[0];
    const float* loss = (const float*)d_in[1];
    const float* b11  = (const float*)d_in[2];
    const float* b12  = (const float*)d_in[3];
    const float* b13  = (const float*)d_in[4];
    const float* b21  = (const float*)d_in[5];
    const float* b22  = (const float*)d_in[6];
    const float* b23  = (const float*)d_in[7];
    const float* w1   = (const float*)d_in[8];
    const float* w2   = (const float*)d_in[9];
    const float* g1   = (const float*)d_in[10];
    const float* be1  = (const float*)d_in[11];
    const float* m1   = (const float*)d_in[12];
    const float* v1   = (const float*)d_in[13];
    const float* g2   = (const float*)d_in[14];
    const float* be2  = (const float*)d_in[15];
    const float* m2   = (const float*)d_in[16];
    const float* v2   = (const float*)d_in[17];
    const float* a1   = (const float*)d_in[18];
    const float* a2   = (const float*)d_in[19];

    char* ws = (char*)d_ws;
    signed char* act1 = (signed char*)(ws + WS_ACT1);
    signed char* act2 = (signed char*)(ws + WS_ACT2);
    float*       out1 = (float*)(ws + WS_OUT1);
    signed char* pB1  = (signed char*)(ws + WS_PB1);
    signed char* pB2  = (signed char*)(ws + WS_PB2);
    float* s1 = (float*)(ws + WS_S1);
    float* c1 = (float*)(ws + WS_C1);
    float* s2 = (float*)(ws + WS_S2);
    float* c2 = (float*)(ws + WS_C2);
    float* zb = (float*)(ws + WS_ZBUF);

    float* out = (float*)d_out;
    float* outLoss = out + (size_t)NPIX * 256;

    prep_channels<<<1, 256, 0, stream>>>(w1, w2, g1, be1, m1, v1,
                                         g2, be2, m2, v2, loss,
                                         s1, c1, s2, c2, zb, outLoss);
    pack_weights<<<640, 256, 0, stream>>>(w1, w2, pB1, pB2);
    act1_k<<<6272, 256, 0, stream>>>(x, b11, act1);
    conv3x3_wmma<<<dim3(196, 2), 256, 0, stream>>>(
        act1, pB1, x, s1, c1, b12, a1, b13, b21, out1, act2,
        (const signed char*)zb);
    conv1x1_wmma<<<dim3(196, 2), 256, 0, stream>>>(
        act2, pB2, out1, s2, c2, b22, a2, b23, out);
}
